// OptPosEncVol_51281909514409
// MI455X (gfx1250) — matrix operand order repaired and checked by reading, not compiled, not run on hardware
//
#include <hip/hip_runtime.h>
#include <hip/hip_bf16.h>

typedef __attribute__((ext_vector_type(16))) _Float16 v16h;
typedef __attribute__((ext_vector_type(8)))  float    v8f;

#define CODE_NUM  64
#define TBL       (CODE_NUM * CODE_NUM * CODE_NUM)   // 262144
#define NCHAN     64
#define WAVES_PER_BLOCK 8

// ---------------------------------------------------------------------------
// Pass 1: transpose [C=64, T] f32 table -> [T, C=64] f16 table in workspace.
// Makes the per-(point,corner) gather a 128B contiguous run of all 64
// channels (vs 64 separate 1MB-strided lines in the source layout) and moves
// the f32->f16 conversion out of the hot gather loop.
// ---------------------------------------------------------------------------
__global__ __launch_bounds__(256)
void transpose_code_f16(const float* __restrict__ src, _Float16* __restrict__ dst,
                        int T)
{
    __shared__ _Float16 tile[64][72];          // padded vs bank conflicts
    const int t0   = blockIdx.x * 64;
    const int tid  = threadIdx.x;
    const int lo6  = tid & 63;
    const int hi2  = tid >> 6;                 // 0..3

#pragma unroll
    for (int r = 0; r < 16; ++r) {             // coalesced reads along T
        const int c = hi2 + r * 4;
        tile[lo6][c] = (_Float16)src[(size_t)c * T + t0 + lo6];
    }
    __syncthreads();
#pragma unroll
    for (int r = 0; r < 16; ++r) {             // coalesced writes along C
        const int t = hi2 + r * 4;
        dst[(size_t)(t0 + t) * NCHAN + lo6] = tile[t][lo6];
    }
}

// ---------------------------------------------------------------------------
// Pass 2: one wave32 per tile of 16 points x 64 channels.
// v_wmma_f32_16x16x32_f16: M = 16 points, N = 16 channels, K = 32 =
// (4 points x 8 corners) block-diagonal. 4 accumulating WMMAs cover the 16
// points, 4 channel groups cover 64 channels -> 16 WMMAs / tile.
// All gathers use a wave-uniform base + 32-bit per-lane element index so the
// compiler can emit the native SGPR-base + VGPR-offset global-load form.
// ---------------------------------------------------------------------------
template <bool USET>
__global__ __launch_bounds__(256)
void OptPosEncVol_trilinear_wmma(const float*    __restrict__ coords,
                                 const float*    __restrict__ code,     // [C, T]
                                 const _Float16* __restrict__ codeT,    // [T, C]
                                 const int*      __restrict__ idx_scalar,
                                 float*          __restrict__ out,
                                 int npts)
{
    __shared__ int sIdx[WAVES_PER_BLOCK][16][8];   // per-wave corner indices

    const int lane = threadIdx.x & 31;
    const int wave = threadIdx.x >> 5;
    const int m    = lane & 15;     // point-in-tile (A/D row) == channel-in-group (B/D col)
    const int hi   = lane >> 4;     // wave half

    const int tile = blockIdx.x * WAVES_PER_BLOCK + wave;
    const int p0   = tile * 16;
    if (p0 >= npts) return;         // wave-uniform; EXEC stays all-ones below
    const int p    = p0 + m;        // this lane's point (both halves compute it)

    const int off = idx_scalar[0] * TBL;

    // ---- per-point corner indices + trilinear weights -----------------------
    const float cx = (coords[3 * p + 0] + 1.0f) * ((CODE_NUM - 1) * 0.5f);
    const float cy = (coords[3 * p + 1] + 1.0f) * ((CODE_NUM - 1) * 0.5f);
    const float cz = (coords[3 * p + 2] + 1.0f) * ((CODE_NUM - 1) * 0.5f);
    const float fx = floorf(cx), fy = floorf(cy), fz = floorf(cz);
    const float dx = cx - fx,    dy = cy - fy,    dz = cz - fz;
    const int   i0 = (int)fx + CODE_NUM * (int)fy + CODE_NUM * CODE_NUM * (int)fz + off;
    const float ex = 1.0f - dx, ey = 1.0f - dy, ez = 1.0f - dz;

    _Float16 wh[8];
    int      ci[8];
#pragma unroll
    for (int k = 0; k < 8; ++k) {
        // reference corner order: mask[k] = ((k>>2)&1,(k>>1)&1,k&1), mul = (1,64,4096)
        const int bx = (k >> 2) & 1, by = (k >> 1) & 1, bz = k & 1;
        const float w = (bx ? dx : ex) * (by ? dy : ey) * (bz ? dz : ez);
        wh[k] = (_Float16)w;
        ci[k] = i0 + bx + by * CODE_NUM + bz * CODE_NUM * CODE_NUM;
    }

    // share corner indices across the wave (LDS in-order within a wave)
    if (hi == 0) {
#pragma unroll
        for (int k = 0; k < 8; ++k) sIdx[wave][m][k] = ci[k];
    }
    asm volatile("" ::: "memory");
    __builtin_amdgcn_wave_barrier();
    asm volatile("" ::: "memory");

    v8f acc[4] = {v8f{}, v8f{}, v8f{}, v8f{}};
    const _Float16 zh = (_Float16)0.0f;

#pragma unroll
    for (int g = 0; g < 4; ++g) {
        // ---- A_g: block-diagonal weights (ISA 16-bit A 16x32 layout) --------
        // h[0..7]  hold K = 8*hi    + 0..7  -> K>>3 == hi
        // h[8..15] hold K = 16+8*hi + 0..7  -> K>>3 == 2+hi
        const bool lo = (m == 4 * g + hi);
        const bool up = (m == 4 * g + 2 + hi);
        v16h A;
#pragma unroll
        for (int t = 0; t < 8; ++t) {
            A[t]     = lo ? wh[t] : zh;
            A[8 + t] = up ? wh[t] : zh;
        }

        // corner indices of the two points this lane gathers for B;
        // pre-scale to 32-bit row-base element offsets (shared by all 4 cg)
        const int pa = 4 * g + hi;
        int rowA[8], rowB[8];
#pragma unroll
        for (int t = 0; t < 8; ++t) {
            rowA[t] = sIdx[wave][pa][t]     * NCHAN;   // element index, < 2^25
            rowB[t] = sIdx[wave][pa + 2][t] * NCHAN;
        }

#pragma unroll
        for (int cg = 0; cg < 4; ++cg) {
            const int ch = 16 * cg + m;
            v16h B;
            if constexpr (USET) {
                // uniform base + 32-bit per-lane index -> GVS addressing;
                // half-wave reads a contiguous 32B run per (point,corner)
#pragma unroll
                for (int t = 0; t < 8; ++t) {
                    B[t]     = codeT[rowA[t] + ch];
                    B[8 + t] = codeT[rowB[t] + ch];
                }
            } else {
                // fallback: gather from the original [C, T] f32 layout,
                // still uniform-base 32-bit indexing (ch*TBL+idx < 2^24)
                const int cb = ch * TBL;
#pragma unroll
                for (int t = 0; t < 8; ++t) {
                    B[t]     = (_Float16)code[cb + rowA[t] / NCHAN];
                    B[8 + t] = (_Float16)code[cb + rowB[t] / NCHAN];
                }
            }
            acc[cg] = __builtin_amdgcn_wmma_f32_16x16x32_f16(
                /*neg_a=*/false, A, /*neg_b=*/false, B,
                /*c_mod=*/(short)0, acc[cg],
                /*reuse_a=*/false, /*reuse_b=*/false);
        }
    }

    // ---- store: C/D layout -> lane holds (row r + 8*hi, col m) --------------
#pragma unroll
    for (int cg = 0; cg < 4; ++cg) {
#pragma unroll
        for (int r = 0; r < 8; ++r) {
            const int row = p0 + r + 8 * hi;
            out[(size_t)row * NCHAN + 16 * cg + m] = acc[cg][r];
        }
    }
}

extern "C" void kernel_launch(void* const* d_in, const int* in_sizes, int n_in,
                              void* d_out, int out_size, void* d_ws, size_t ws_size,
                              hipStream_t stream) {
    const float* coords = (const float*)d_in[0];
    const float* code   = (const float*)d_in[1];
    const int*   idx    = (const int*)d_in[2];
    float*       out    = (float*)d_out;

    const int npts   = in_sizes[0] / 3;                               // 524288
    const int tiles  = (npts + 15) / 16;                              // 32768
    const int blocks = (tiles + WAVES_PER_BLOCK - 1) / WAVES_PER_BLOCK; // 4096

    const int    T       = in_sizes[1] / NCHAN;                       // 262144
    const size_t t_bytes = (size_t)T * NCHAN * sizeof(_Float16);      // 32 MB

    if (ws_size >= t_bytes && (T % 64) == 0) {
        _Float16* codeT = (_Float16*)d_ws;
        transpose_code_f16<<<T / 64, 256, 0, stream>>>(code, codeT, T);
        OptPosEncVol_trilinear_wmma<true><<<blocks, 256, 0, stream>>>(
            coords, code, codeT, idx, out, npts);
    } else {
        OptPosEncVol_trilinear_wmma<false><<<blocks, 256, 0, stream>>>(
            coords, code, nullptr, idx, out, npts);
    }
}
